// Attention_37606733643961
// MI455X (gfx1250) — matrix operand order repaired and checked by reading, not compiled, not run on hardware
//
#include <hip/hip_runtime.h>
#include <stdint.h>

typedef __attribute__((ext_vector_type(16))) __bf16 v16bf;
typedef __attribute__((ext_vector_type(8)))  float  v8f;

#define S_LEN   3136
#define DIM     256
#define NHEADS  8
#define HDIM    32
#define NPAIRS  32        // B * NHEADS
#define MROWS   12544     // B * S
#define QK_SCALE 0.4204482076268573f   // 32^-0.25, applied to both q and k

union Frag {
  v16bf  v;
  __bf16 e[16];
  uint4  u4[2];
};

__device__ inline v8f wmma_bf16(v16bf a, v16bf b, v8f c) {
  return __builtin_amdgcn_wmma_f32_16x16x32_bf16(false, a, false, b, (short)0, c,
                                                 false, false);
}

// A-matrix fragment (16x32, row m = lane%16). Per-lane K runs: [h*8, h*8+8) and
// [16+h*8, 16+h*8+8)  (ISA 7.12.2, 16-bit A 16x32 table).
__device__ inline v16bf load_a_bf16(const __bf16* rowptr, int h) {
  Frag f;
  f.u4[0] = *(const uint4*)(rowptr + h * 8);
  f.u4[1] = *(const uint4*)(rowptr + 16 + h * 8);
  return f.v;
}

__device__ inline v16bf load_a_f32(const float* rowptr, int h) {
  float t[16];
  *(float4*)&t[0]  = *(const float4*)(rowptr + h * 8);
  *(float4*)&t[4]  = *(const float4*)(rowptr + h * 8 + 4);
  *(float4*)&t[8]  = *(const float4*)(rowptr + 16 + h * 8);
  *(float4*)&t[12] = *(const float4*)(rowptr + 16 + h * 8 + 4);
  Frag f;
#pragma unroll
  for (int i = 0; i < 16; ++i) f.e[i] = (__bf16)t[i];
  return f.v;
}

// B-matrix fragment (32x16, col n = lane%16). Per-lane K runs: [h*8, h*8+8) and
// [16+h*8, 16+h*8+8), matching the A fragment. colptr points at (n, k0) in a
// [N][K]-contiguous buffer.
__device__ inline v16bf load_b_bf16(const __bf16* colptr, int h) {
  Frag f;
  f.u4[0] = *(const uint4*)(colptr + h * 8);
  f.u4[1] = *(const uint4*)(colptr + 16 + h * 8);
  return f.v;
}

// Butterfly reductions across the 16 lanes of a half-wave (row of a C tile).
__device__ inline float redmax16(float x) {
#pragma unroll
  for (int off = 1; off < 16; off <<= 1) x = fmaxf(x, __shfl_xor(x, off, 32));
  return x;
}
__device__ inline float redsum16(float x) {
#pragma unroll
  for (int off = 1; off < 16; off <<= 1) x += __shfl_xor(x, off, 32);
  return x;
}

// W[K][N] fp32  ->  Wt[N][K] bf16
__global__ void transpose_cvt_kernel(const float* __restrict__ W,
                                     __bf16* __restrict__ Wt, int K, int N) {
  int idx = blockIdx.x * blockDim.x + threadIdx.x;
  if (idx >= K * N) return;
  int k = idx / N, n = idx - k * N;
  Wt[(size_t)n * K + k] = (__bf16)W[idx];
}

// qkv = x @ w_qkv ; wave computes 16 rows x 128 cols (8 wmma tiles); scatter
// into Q,K (scaled, [pair][S][32]) and V^T ([pair][32][S]).
__global__ void __launch_bounds__(128)
qkv_gemm_kernel(const float* __restrict__ X, const __bf16* __restrict__ Wt,
                __bf16* __restrict__ Qb, __bf16* __restrict__ Kb,
                __bf16* __restrict__ Vt) {
  const int lane = threadIdx.x & 31, w = threadIdx.x >> 5;
  const int h = lane >> 4, ln = lane & 15;
  const int row0 = blockIdx.x * 64 + w * 16;
  const int n0 = blockIdx.y * 128;
  v8f acc[8];
#pragma unroll
  for (int t = 0; t < 8; ++t)
#pragma unroll
    for (int r = 0; r < 8; ++r) acc[t][r] = 0.f;

  const float* arow = X + (size_t)(row0 + ln) * DIM;
#pragma unroll
  for (int k0 = 0; k0 < DIM; k0 += 32) {
    v16bf af = load_a_f32(arow + k0, h);
#pragma unroll
    for (int t = 0; t < 8; ++t) {
      v16bf bf = load_b_bf16(Wt + (size_t)(n0 + t * 16 + ln) * DIM + k0, h);
      acc[t] = wmma_bf16(af, bf, acc[t]);
    }
  }

  const int b = row0 / S_LEN;          // 3136 % 64 == 0: whole tile in one image
  const int s0 = row0 - b * S_LEN;
#pragma unroll
  for (int t = 0; t < 8; ++t) {
    const int d = n0 + t * 16 + ln;             // column in [0,768)
    const int which = d >> 8;                   // 0=q 1=k 2=v
    const int head = (d >> 5) & 7;
    const int hd = d & 31;
    const int pair = b * NHEADS + head;
#pragma unroll
    for (int r = 0; r < 8; ++r) {
      const int s = s0 + r + 8 * h;             // C layout: M = r + 8*half
      const float val = acc[t][r];
      if (which == 0)
        Qb[((size_t)pair * S_LEN + s) * HDIM + hd] = (__bf16)(val * QK_SCALE);
      else if (which == 1)
        Kb[((size_t)pair * S_LEN + s) * HDIM + hd] = (__bf16)(val * QK_SCALE);
      else
        Vt[((size_t)pair * HDIM + hd) * S_LEN + s] = (__bf16)val;
    }
  }
}

// Flash attention: one wave = 16 queries of one (batch,head) pair.
// Software-pipelined: next K tile + current V tile loads are issued before the
// softmax VALU block so load latency hides under exp/shuffle work.
__global__ void __launch_bounds__(128)
flash_attn_kernel(const __bf16* __restrict__ Qb, const __bf16* __restrict__ Kb,
                  const __bf16* __restrict__ Vt, float* __restrict__ O) {
  __shared__ __align__(16) __bf16 plds[4][16][32];   // per-wave P tile (C->A relayout)
  const int lane = threadIdx.x & 31, w = threadIdx.x >> 5;
  const int h = lane >> 4, ln = lane & 15;
  const int pair = blockIdx.y;
  const int q0 = blockIdx.x * 64 + w * 16;

  const __bf16* Qp = Qb + (size_t)pair * S_LEN * HDIM;
  const __bf16* Kp = Kb + (size_t)pair * S_LEN * HDIM;
  const __bf16* Vp = Vt + (size_t)pair * HDIM * S_LEN;

  const v16bf qf = load_a_bf16(Qp + (size_t)(q0 + ln) * HDIM, h);  // 16x32 Q tile
  const v8f zero = {0.f, 0.f, 0.f, 0.f, 0.f, 0.f, 0.f, 0.f};
  v8f o0, o1;                    // O accumulator, hd 0-15 and 16-31
  float mrow[8], lrow[8];        // row stats for rows m = r + 8*h (lane-local!)
#pragma unroll
  for (int r = 0; r < 8; ++r) { o0[r] = 0.f; o1[r] = 0.f; mrow[r] = -1e30f; lrow[r] = 0.f; }

  // prologue: K fragments for tile 0
  v16bf kf0 = load_b_bf16(Kp + (size_t)(0 + ln) * HDIM, h);
  v16bf kf1 = load_b_bf16(Kp + (size_t)(16 + ln) * HDIM, h);

  for (int k0 = 0; k0 < S_LEN; k0 += 32) {
    // scores: Q(16x32) x K^T(32x16) — head_dim is exactly the WMMA K dim
    v8f s0 = wmma_bf16(qf, kf0, zero);
    v8f s1 = wmma_bf16(qf, kf1, zero);

    // issue next-tile K loads and current V loads NOW; first use is after the
    // softmax block, so latency overlaps with exp/butterfly VALU work.
    const int kn = (k0 + 32 < S_LEN) ? (k0 + 32) : 0;
    v16bf nkf0 = load_b_bf16(Kp + (size_t)(kn + ln) * HDIM, h);
    v16bf nkf1 = load_b_bf16(Kp + (size_t)(kn + 16 + ln) * HDIM, h);
    v16bf vf0 = load_b_bf16(Vp + (size_t)ln * S_LEN + k0, h);
    v16bf vf1 = load_b_bf16(Vp + (size_t)(16 + ln) * S_LEN + k0, h);

    // online softmax; C-layout row m = r + 8*h lives in this lane's slot r
#pragma unroll
    for (int r = 0; r < 8; ++r) {
      float mx = redmax16(fmaxf(s0[r], s1[r]));
      float mnew = fmaxf(mrow[r], mx);
      float corr = __expf(mrow[r] - mnew);
      float e0 = __expf(s0[r] - mnew);
      float e1 = __expf(s1[r] - mnew);
      lrow[r] = lrow[r] * corr + redsum16(e0 + e1);
      mrow[r] = mnew;
      o0[r] *= corr;
      o1[r] *= corr;
      plds[w][r + 8 * h][ln] = (__bf16)e0;        // P in [q][key_local] layout
      plds[w][r + 8 * h][16 + ln] = (__bf16)e1;
    }

    // re-read P as an A fragment (16 queries x 32 keys); wave-private LDS,
    // DS ops are in-order within a wave.
    Frag pf;
    pf.u4[0] = *(const uint4*)&plds[w][ln][h * 8];
    pf.u4[1] = *(const uint4*)&plds[w][ln][16 + h * 8];

    // O += P(16x32) x V(32keys x 32hd); V^T rows give contiguous B runs
    o0 = wmma_bf16(pf.v, vf0, o0);
    o1 = wmma_bf16(pf.v, vf1, o1);

    kf0 = nkf0;
    kf1 = nkf1;
  }

  const int b = pair >> 3, head = pair & 7;
  float* obase = O + (size_t)(b * S_LEN + q0) * DIM + head * HDIM;
#pragma unroll
  for (int r = 0; r < 8; ++r) {
    float inv = 1.0f / lrow[r];
    float* orow = obase + (size_t)(r + 8 * h) * DIM;
    orow[ln] = o0[r] * inv;
    orow[16 + ln] = o1[r] * inv;
  }
}

// out = attn_out @ w_proj ; wave computes 16 rows x 128 cols
__global__ void __launch_bounds__(128)
proj_gemm_kernel(const float* __restrict__ A, const __bf16* __restrict__ Wt,
                 float* __restrict__ Out) {
  const int lane = threadIdx.x & 31, w = threadIdx.x >> 5;
  const int h = lane >> 4, ln = lane & 15;
  const int row0 = blockIdx.x * 64 + w * 16;
  const int n0 = blockIdx.y * 128;
  v8f acc[8];
#pragma unroll
  for (int t = 0; t < 8; ++t)
#pragma unroll
    for (int r = 0; r < 8; ++r) acc[t][r] = 0.f;

  const float* arow = A + (size_t)(row0 + ln) * DIM;
#pragma unroll
  for (int k0 = 0; k0 < DIM; k0 += 32) {
    v16bf af = load_a_f32(arow + k0, h);
#pragma unroll
    for (int t = 0; t < 8; ++t) {
      v16bf bf = load_b_bf16(Wt + (size_t)(n0 + t * 16 + ln) * DIM + k0, h);
      acc[t] = wmma_bf16(af, bf, acc[t]);
    }
  }
#pragma unroll
  for (int t = 0; t < 8; ++t)
#pragma unroll
    for (int r = 0; r < 8; ++r)
      Out[(size_t)(row0 + r + 8 * h) * DIM + n0 + t * 16 + ln] = acc[t][r];
}

extern "C" void kernel_launch(void* const* d_in, const int* in_sizes, int n_in,
                              void* d_out, int out_size, void* d_ws, size_t ws_size,
                              hipStream_t stream) {
  const float* x = (const float*)d_in[0];        // [4,56,56,256]
  const float* w_qkv = (const float*)d_in[1];    // [256,768]
  const float* w_proj = (const float*)d_in[2];   // [256,256]
  float* out = (float*)d_out;                    // [4,56,56,256]

  char* ws = (char*)d_ws;
  size_t off = 0;
  auto carve = [&](size_t bytes) {
    void* p = ws + off;
    off = (off + bytes + 255) & ~(size_t)255;
    return p;
  };
  __bf16* Qb = (__bf16*)carve((size_t)NPAIRS * S_LEN * HDIM * 2);   // 6.4 MB
  __bf16* Kb = (__bf16*)carve((size_t)NPAIRS * S_LEN * HDIM * 2);   // 6.4 MB
  __bf16* Vt = (__bf16*)carve((size_t)NPAIRS * HDIM * S_LEN * 2);   // 6.4 MB
  __bf16* WqkvT = (__bf16*)carve((size_t)DIM * 768 * 2);
  __bf16* WprojT = (__bf16*)carve((size_t)DIM * DIM * 2);
  float* AttnO = (float*)carve((size_t)MROWS * DIM * 4);            // 12.8 MB

  transpose_cvt_kernel<<<(DIM * 768 + 255) / 256, 256, 0, stream>>>(w_qkv, WqkvT, DIM, 768);
  transpose_cvt_kernel<<<(DIM * DIM + 255) / 256, 256, 0, stream>>>(w_proj, WprojT, DIM, DIM);
  qkv_gemm_kernel<<<dim3(MROWS / 64, 768 / 128), 128, 0, stream>>>(x, WqkvT, Qb, Kb, Vt);
  flash_attn_kernel<<<dim3(S_LEN / 64, NPAIRS), 128, 0, stream>>>(Qb, Kb, Vt, AttnO);
  proj_gemm_kernel<<<dim3(MROWS / 64, DIM / 128), 128, 0, stream>>>(AttnO, WprojT, out);
}
